// RotarySelfAttention_13151189861180
// MI455X (gfx1250) — compile-verified, matrix operand order checked by reading, not run on hardware
//
#include <hip/hip_runtime.h>
#include <hip/hip_bf16.h>
#include <float.h>

// ---- problem dims ----
#define B_  8
#define TN_ 512
#define TP_ 2048
#define D_  1024
#define H_  16
#define DH_ 64
#define TK_ (TP_ + TN_)   // 2560

typedef __attribute__((ext_vector_type(16))) __bf16 v16bf;
typedef __attribute__((ext_vector_type(8)))  float  v8f;

union FragU {
    v16bf v;
    uint4 q[2];
    unsigned short u[16];
};

__device__ __forceinline__ unsigned short f2bf(float f) {
    unsigned u = __float_as_uint(f);
    u += 0x7fffu + ((u >> 16) & 1u);   // round-to-nearest-even
    return (unsigned short)(u >> 16);
}

__device__ __forceinline__ float redmax16(float v) {
    v = fmaxf(v, __shfl_xor(v, 1, 32));
    v = fmaxf(v, __shfl_xor(v, 2, 32));
    v = fmaxf(v, __shfl_xor(v, 4, 32));
    v = fmaxf(v, __shfl_xor(v, 8, 32));
    return v;
}
__device__ __forceinline__ float redsum16(float v) {
    v += __shfl_xor(v, 1, 32);
    v += __shfl_xor(v, 2, 32);
    v += __shfl_xor(v, 4, 32);
    v += __shfl_xor(v, 8, 32);
    return v;
}

// ---------- helper kernels ----------
__global__ void total_len_kernel(const int* __restrict__ plen,
                                 const unsigned char* __restrict__ vmask,
                                 int* __restrict__ out_tail, int* __restrict__ ws_tl) {
    int b = threadIdx.x;
    if (b >= B_) return;
    int s = 0;
    for (int t = 0; t < TN_; ++t) s += vmask[b * TN_ + t] ? 1 : 0;
    int tot = plen[b] + s;
    out_tail[b] = tot;
    ws_tl[b]    = tot;
}

__global__ void cvt_f32_bf16(const float* __restrict__ in,
                             unsigned short* __restrict__ out, int n) {
    int i = blockIdx.x * blockDim.x + threadIdx.x;
    if (i < n) out[i] = f2bf(in[i]);
}

// Wt[n][k] = W[k][n]  (B-operand fragments then load contiguously along K)
__global__ void cvt_w_transpose(const float* __restrict__ W,
                                unsigned short* __restrict__ Wt) {
    int i = blockIdx.x * blockDim.x + threadIdx.x;   // i = n*1024 + k
    int n = i >> 10, k = i & (D_ - 1);
    Wt[i] = f2bf(W[k * D_ + n]);
}

// Past cache -> unified bf16 buffers, zeroed beyond past_len.
//   kt: [B,H,TK,DH]  (rows contiguous in d)   -- fills j in [0,TP)
//   vt: [B,H,DH,TK]  (transposed: cols contiguous in key) -- fills j in [0,TP)
// V tile is staged global->LDS with the CDNA5 async-DMA path (ASYNCcnt),
// overlapped with the K-side conversion, then transposed out of LDS.
#define VROW_ 72   // padded LDS row (floats) to avoid bank conflicts on transpose
__global__ __launch_bounds__(256) void cvt_past_kernel(
        const float* __restrict__ pastK, const float* __restrict__ pastV,
        const int* __restrict__ plen,
        unsigned short* __restrict__ kt, unsigned short* __restrict__ vt) {
    __shared__ __align__(16) float tilef[16][VROW_];
    const int blk = blockIdx.x;              // bh*(TP/16) + jt
    const int jt  = blk & (TP_ / 16 - 1);    // 128 tiles of 16 keys
    const int bh  = blk >> 7;
    const int j0  = jt * 16;
    const int pl  = plen[bh >> 4];
    const int t   = threadIdx.x;

    // ---- async DMA: 16x64 f32 V tile -> LDS (one b128 per thread) ----
    {
        int j = t >> 4;                      // row 0..15
        int ch = t & 15;                     // 16B chunk within row (4 floats)
        unsigned lds_off = (unsigned)(unsigned long long)(&tilef[0][0])
                         + (unsigned)(j * VROW_ * 4 + ch * 16);
        unsigned long long ga = (unsigned long long)
            (pastV + ((size_t)bh * TP_ + j0 + j) * DH_ + ch * 4);
        asm volatile("global_load_async_to_lds_b128 %0, %1, off"
                     :: "v"(lds_off), "v"(ga) : "memory");
    }

    // ---- overlapped: K conversion straight from global ----
#pragma unroll
    for (int p = 0; p < 4; ++p) {
        int e = t + p * 256;                 // 0..1023 : 16 keys x 64 d
        int j = e >> 6, d = e & 63;
        int jj = j0 + j;
        size_t src = ((size_t)bh * TP_ + jj) * DH_ + d;
        unsigned short kb = (jj < pl) ? f2bf(pastK[src]) : (unsigned short)0;
        kt[((size_t)bh * TK_ + jj) * DH_ + d] = kb;
    }

    // ---- wait for our wave's async transfers, publish across the block ----
    asm volatile("s_wait_asynccnt 0" ::: "memory");
    __syncthreads();

    // ---- transpose + mask + convert out of LDS ----
#pragma unroll
    for (int p = 0; p < 4; ++p) {
        int e = t + p * 256;
        int d = e >> 4, j = e & 15;          // consecutive threads -> consecutive keys
        int jj = j0 + j;
        unsigned short vb = (jj < pl) ? f2bf(tilef[j][d]) : (unsigned short)0;
        vt[((size_t)bh * DH_ + d) * TK_ + jj] = vb;
    }
}

// RoPE + validity mask + bf16 pack.  q -> [B,H,TN,DH]; k -> kt[.., TP+t, d]; v -> vt[.., d, TP+t]
__global__ void rope_cvt(const float* __restrict__ qf, const float* __restrict__ kf,
                         const float* __restrict__ vf,
                         const int* __restrict__ plen, const unsigned char* __restrict__ vmask,
                         const float* __restrict__ inv_freq,
                         unsigned short* __restrict__ qb, unsigned short* __restrict__ kt,
                         unsigned short* __restrict__ vt) {
    int i   = blockIdx.x * blockDim.x + threadIdx.x;  // over B*TN*D
    int col = i & (D_ - 1);
    int tok = i >> 10;                 // b*TN + t
    int b   = tok >> 9, t = tok & (TN_ - 1);
    int d   = col & (DH_ - 1), h = col >> 6;
    int bh  = b * H_ + h;
    float m   = vmask[tok] ? 1.0f : 0.0f;
    float pos = vmask[tok] ? (float)(plen[b] + t) : 0.0f;
    float ang = pos * inv_freq[d & 31];
    float cs = __cosf(ang), sn = __sinf(ang);
    float qv = qf[i], kv = kf[i];
    float qp = (d < 32) ? -qf[i + 32] : qf[i - 32];   // rotate_half
    float kp = (d < 32) ? -kf[i + 32] : kf[i - 32];
    qb[((size_t)bh * TN_ + t) * DH_ + d]        = f2bf((qv * cs + qp * sn) * m);
    kt[((size_t)bh * TK_ + TP_ + t) * DH_ + d]  = f2bf((kv * cs + kp * sn) * m);
    vt[((size_t)bh * DH_ + d) * TK_ + TP_ + t]  = f2bf(vf[i] * m);
}

// ---------- WMMA GEMM: C[M][N] = A[M][K](bf16) * Bt[N][K](bf16)^T + bias ----------
__global__ __launch_bounds__(256) void gemm_wmma_bf16(
        const unsigned short* __restrict__ A,
        const unsigned short* __restrict__ Bt,
        const float* __restrict__ bias,
        const unsigned char* __restrict__ rowmask,
        float* __restrict__ C, int M, int N, int K) {
    const int lane = threadIdx.x & 31;
    const int wid  = blockIdx.x * (blockDim.x >> 5) + (threadIdx.x >> 5);
    const int tilesN = N >> 4;
    const int tm = wid / tilesN, tn = wid % tilesN;
    if (tm * 16 >= M) return;                 // wave-uniform
    const int c = lane & 15, g = lane >> 4;

    const unsigned short* arow = A  + (size_t)(tm * 16 + c) * K;
    const unsigned short* brow = Bt + (size_t)(tn * 16 + c) * K;
    v8f acc = {};
    for (int kk = 0; kk < K; kk += 32) {
        FragU a, b;
        a.q[0] = *(const uint4*)(arow + kk + g * 8);
        a.q[1] = *(const uint4*)(arow + kk + 16 + g * 8);
        b.q[0] = *(const uint4*)(brow + kk + g * 8);
        b.q[1] = *(const uint4*)(brow + kk + 16 + g * 8);
        acc = __builtin_amdgcn_wmma_f32_16x16x32_bf16(false, a.v, false, b.v,
                                                      (short)0, acc, false, false);
    }
    const int coln = tn * 16 + c;
    float bvv = bias ? bias[coln] : 0.0f;
#pragma unroll
    for (int r = 0; r < 8; ++r) {
        int row = tm * 16 + g * 8 + r;
        float v = acc[r] + bvv;
        if (rowmask) v = rowmask[row] ? v : 0.0f;
        C[(size_t)row * N + coln] = v;
    }
}

// ---------- flash attention: one wave = one 16-query tile of one (b,h) ----------
// kt: [B,H,TK,DH] bf16 (pre-masked).  vt: [B,H,DH,TK] bf16 (pre-masked, transposed).
__global__ __launch_bounds__(256) void attn_flash_wmma(
        const unsigned short* __restrict__ qb, const unsigned short* __restrict__ kt,
        const unsigned short* __restrict__ vt,
        const int* __restrict__ tlen,
        unsigned short* __restrict__ attn_out /* [B*TN][D] bf16 */) {
    __shared__ __align__(16) unsigned short plds[8][16 * 32];
    const int lane = threadIdx.x & 31;
    const int wib  = threadIdx.x >> 5;
    const int wid  = blockIdx.x * 8 + wib;       // 0 .. B*H*(TN/16)-1 = 4095
    const int qt = wid & 31;
    const int bh = wid >> 5;                     // b*H + h
    const int b  = bh >> 4, h = bh & 15;
    const int c  = lane & 15, g = lane >> 4;
    const int q0 = qt * 16;

    const int tl = tlen[b];

    // Q fragments (A operand): rows q0..q0+15, d in [0,32) / [32,64)
    const unsigned short* qrow = qb + ((size_t)bh * TN_ + q0 + c) * DH_;
    FragU fq0, fq1;
    fq0.q[0] = *(const uint4*)(qrow + g * 8);
    fq0.q[1] = *(const uint4*)(qrow + 16 + g * 8);
    fq1.q[0] = *(const uint4*)(qrow + 32 + g * 8);
    fq1.q[1] = *(const uint4*)(qrow + 48 + g * 8);

    // per-lane base pointers (pure pointer arithmetic in the loop)
    const unsigned short* kbase = kt + ((size_t)bh * TK_ + c) * DH_;       // key row c
    const unsigned short* vbase = vt + ((size_t)bh * DH_ + c) * TK_;       // d-col c

    v8f o[4] = {};
    float m_r[8], l_r[8];
#pragma unroll
    for (int r = 0; r < 8; ++r) { m_r[r] = -FLT_MAX; l_r[r] = 0.0f; }

    for (int cb = 0; cb < TK_; cb += 32) {
        // prefetch next chunk's K rows and V columns (gfx1250 global_prefetch)
        if (cb + 32 < TK_) {
            __builtin_prefetch(kbase + (size_t)(cb + 32) * DH_, 0, 0);
            __builtin_prefetch(kbase + (size_t)(cb + 48) * DH_, 0, 0);
            __builtin_prefetch(vbase + cb + 32, 0, 0);
        }
        // ---- scores for two 16-key tiles ----
        v8f s01[2];
#pragma unroll
        for (int t = 0; t < 2; ++t) {
            const unsigned short* kr = kbase + (size_t)(cb + t * 16) * DH_;
            FragU k0, k1;
            k0.q[0] = *(const uint4*)(kr + g * 8);
            k0.q[1] = *(const uint4*)(kr + 16 + g * 8);
            k1.q[0] = *(const uint4*)(kr + 32 + g * 8);
            k1.q[1] = *(const uint4*)(kr + 48 + g * 8);
            v8f s = {};
            s = __builtin_amdgcn_wmma_f32_16x16x32_bf16(false, fq0.v, false, k0.v,
                                                        (short)0, s, false, false);
            s = __builtin_amdgcn_wmma_f32_16x16x32_bf16(false, fq1.v, false, k1.v,
                                                        (short)0, s, false, false);
            bool valid = (cb + t * 16 + c) < tl;    // key-validity mask
#pragma unroll
            for (int r = 0; r < 8; ++r) s[r] = valid ? s[r] * 0.125f : -FLT_MAX;
            s01[t] = s;
        }
        // ---- online softmax ----
        float alpha[8];
#pragma unroll
        for (int r = 0; r < 8; ++r) {
            float cm = redmax16(fmaxf(s01[0][r], s01[1][r]));
            float mn = fmaxf(m_r[r], cm);
            alpha[r] = __expf(m_r[r] - mn);
            m_r[r] = mn;
        }
#pragma unroll
        for (int t = 0; t < 2; ++t)
#pragma unroll
            for (int r = 0; r < 8; ++r)
                s01[t][r] = __expf(s01[t][r] - m_r[r]);
#pragma unroll
        for (int r = 0; r < 8; ++r) {
            float rs = redsum16(s01[0][r] + s01[1][r]);
            l_r[r] = l_r[r] * alpha[r] + rs;
        }
#pragma unroll
        for (int t = 0; t < 4; ++t)
#pragma unroll
            for (int r = 0; r < 8; ++r) o[t][r] *= alpha[r];

        // ---- re-layout P (C/D layout -> A layout) via LDS ----
#pragma unroll
        for (int t = 0; t < 2; ++t)
#pragma unroll
            for (int r = 0; r < 8; ++r)
                plds[wib][(g * 8 + r) * 32 + t * 16 + c] = f2bf(s01[t][r]);
        FragU pf;
        pf.q[0] = *(const uint4*)&plds[wib][c * 32 + g * 8];
        pf.q[1] = *(const uint4*)&plds[wib][c * 32 + 16 + g * 8];

        // ---- V fragments (B operand, contiguous along keys) and P@V ----
#pragma unroll
        for (int t = 0; t < 4; ++t) {
            const unsigned short* vc = vbase + (size_t)(t * 16) * TK_ + cb;
            FragU vfr;
            vfr.q[0] = *(const uint4*)(vc + g * 8);
            vfr.q[1] = *(const uint4*)(vc + 16 + g * 8);
            o[t] = __builtin_amdgcn_wmma_f32_16x16x32_bf16(false, pf.v, false, vfr.v,
                                                           (short)0, o[t], false, false);
        }
    }
    // ---- finalize: out = o / l, bf16 store into [B*TN][D] (head-concat layout) ----
    float inv_l[8];
#pragma unroll
    for (int r = 0; r < 8; ++r) inv_l[r] = (l_r[r] > 0.0f) ? (1.0f / l_r[r]) : 0.0f;
#pragma unroll
    for (int t = 0; t < 4; ++t)
#pragma unroll
        for (int r = 0; r < 8; ++r) {
            int row = b * TN_ + q0 + g * 8 + r;
            int col = h * DH_ + t * 16 + c;
            attn_out[(size_t)row * D_ + col] = f2bf(o[t][r] * inv_l[r]);
        }
}

// ---------- launcher ----------
extern "C" void kernel_launch(void* const* d_in, const int* in_sizes, int n_in,
                              void* d_out, int out_size, void* d_ws, size_t ws_size,
                              hipStream_t stream) {
    (void)in_sizes; (void)n_in; (void)out_size; (void)ws_size;
    const float* x_new    = (const float*)d_in[0];
    const float* inv_freq = (const float*)d_in[1];
    const float* past_k   = (const float*)d_in[2];
    const float* past_v   = (const float*)d_in[3];
    const int*   plen     = (const int*)d_in[4];
    const unsigned char* vmask = (const unsigned char*)d_in[5];
    const float* Wq = (const float*)d_in[6];  const float* bq = (const float*)d_in[7];
    const float* Wk = (const float*)d_in[8];  const float* bk = (const float*)d_in[9];
    const float* Wv = (const float*)d_in[10]; const float* bv = (const float*)d_in[11];
    const float* Wo = (const float*)d_in[12]; const float* bo = (const float*)d_in[13];
    float* out = (float*)d_out;

    char* ws = (char*)d_ws;
    const size_t MB = 1024u * 1024u;
    unsigned short* xb   = (unsigned short*)(ws + 0);         // 8 MB  x bf16 [4096][1024]
    unsigned short* Wqt  = (unsigned short*)(ws + 8   * MB);  // 2 MB each, transposed bf16
    unsigned short* Wkt  = (unsigned short*)(ws + 10  * MB);
    unsigned short* Wvt  = (unsigned short*)(ws + 12  * MB);
    unsigned short* Wot  = (unsigned short*)(ws + 14  * MB);
    float* qf = (float*)(ws + 16 * MB);                       // 16 MB f32 [4096][1024] each
    float* kf = (float*)(ws + 32 * MB);
    float* vf = (float*)(ws + 48 * MB);
    unsigned short* qbuf = (unsigned short*)(ws + 64  * MB);  // 8 MB  bf16 [B,H,TN,DH]
    unsigned short* ktb  = (unsigned short*)(ws + 72  * MB);  // 40 MB bf16 [B,H,TK,DH]
    unsigned short* vtb  = (unsigned short*)(ws + 112 * MB);  // 40 MB bf16 [B,H,DH,TK]
    unsigned short* abuf = (unsigned short*)(ws + 152 * MB);  // 8 MB  bf16 [4096][1024]
    int* ws_tl = (int*)(ws + 160 * MB);

    int* out_tail = (int*)out + (size_t)B_ * TN_ * D_;

    total_len_kernel<<<1, 32, 0, stream>>>(plen, vmask, out_tail, ws_tl);

    // past KV -> bf16 (K) + bf16 transposed (V), pre-masked   (independent of GEMMs)
    cvt_past_kernel<<<B_ * H_ * (TP_ / 16), 256, 0, stream>>>(past_k, past_v, plen, ktb, vtb);

    const int NX = B_ * TN_ * D_;                 // 4,194,304
    cvt_f32_bf16<<<NX / 256, 256, 0, stream>>>(x_new, xb, NX);
    cvt_w_transpose<<<(D_ * D_) / 256, 256, 0, stream>>>(Wq, Wqt);
    cvt_w_transpose<<<(D_ * D_) / 256, 256, 0, stream>>>(Wk, Wkt);
    cvt_w_transpose<<<(D_ * D_) / 256, 256, 0, stream>>>(Wv, Wvt);
    cvt_w_transpose<<<(D_ * D_) / 256, 256, 0, stream>>>(Wo, Wot);

    // (M/16)*(N/16) = 256*64 = 16384 wave-tiles / 8 waves per block
    gemm_wmma_bf16<<<2048, 256, 0, stream>>>(xb, Wqt, bq, nullptr, qf, B_ * TN_, D_, D_);
    gemm_wmma_bf16<<<2048, 256, 0, stream>>>(xb, Wkt, bk, nullptr, kf, B_ * TN_, D_, D_);
    gemm_wmma_bf16<<<2048, 256, 0, stream>>>(xb, Wvt, bv, nullptr, vf, B_ * TN_, D_, D_);

    rope_cvt<<<NX / 256, 256, 0, stream>>>(qf, kf, vf, plen, vmask, inv_freq,
                                           qbuf, ktb, vtb);

    // B*H*(TN/16) = 4096 waves / 8 per block
    attn_flash_wmma<<<512, 256, 0, stream>>>(qbuf, ktb, vtb, ws_tl, abuf);

    gemm_wmma_bf16<<<2048, 256, 0, stream>>>(abuf, Wot, bo, vmask, out, B_ * TN_, D_, D_);
}
